// dnc_model_58926951301616
// MI455X (gfx1250) — compile-verified, matrix operand order checked by reading, not run on hardware
//
#include <hip/hip_runtime.h>
#include <hip/hip_bf16.h>

// ---- problem constants (from reference) ----
#define T_STEPS 64
#define BATCH   32
#define IN_DIM  64
#define N_MEM   512
#define WD      64
#define RH      4          // read heads
#define HID     256
#define OUT_DIM 64
#define IFACE   471
#define IF_PAD  480        // padded to multiple of 16
#define KG      576        // ctrl_in(320) + h(256)
#define NG      1024       // 4*HID
#define CLIPV   20.0f
#define EPSV    1e-6f

typedef __attribute__((ext_vector_type(16))) _Float16 v16h;
typedef __attribute__((ext_vector_type(8)))  float    v8f;

__device__ __forceinline__ float sigf(float x)   { return 1.0f / (1.0f + __expf(-x)); }
__device__ __forceinline__ float clipf(float x)  { return fminf(fmaxf(x, -CLIPV), CLIPV); }
__device__ __forceinline__ float oneplusf(float x) {
    // 1 + softplus(x), overflow-safe
    return 1.0f + fmaxf(x, 0.0f) + log1pf(__expf(-fabsf(x)));
}

// ---------------- weight conversion: f32 -> f16 transposed ----------------
// WgT  [1024][576] : k<320 from Wx[k][n], else Wh[k-320][n]
// WifT [480][256]  : n<471 from W_if[k][n], else 0
// WoutT[64][512]   : W_out[k][n]
__global__ void convert_weights(const float* __restrict__ Wx,
                                const float* __restrict__ Wh,
                                const float* __restrict__ W_if,
                                const float* __restrict__ W_out,
                                _Float16* __restrict__ WgT,
                                _Float16* __restrict__ WifT,
                                _Float16* __restrict__ WoutT) {
    const int NGE = NG * KG;          // 589824
    const int NIE = IF_PAD * HID;     // 122880
    const int NOE = OUT_DIM * (HID + RH * WD); // 32768
    int i = blockIdx.x * blockDim.x + threadIdx.x;
    if (i < NGE) {
        int n = i / KG, k = i % KG;
        float v = (k < 320) ? Wx[(size_t)k * NG + n] : Wh[(size_t)(k - 320) * NG + n];
        WgT[i] = (_Float16)v;
    } else if (i < NGE + NIE) {
        int j = i - NGE;
        int n = j / HID, k = j % HID;
        WifT[j] = (_Float16)((n < IFACE) ? W_if[(size_t)k * IFACE + n] : 0.0f);
    } else if (i < NGE + NIE + NOE) {
        int j = i - NGE - NIE;
        int n = j / 512, k = j % 512;
        WoutT[j] = (_Float16)W_out[(size_t)k * OUT_DIM + n];
    }
}

// ---------------- WMMA GEMM helper ----------------
// D[16][nTiles*16] = A[16][K] (LDS f16, row stride lda) * W^T, W stored [N][K] f16.
// A-frag per lane: row = lane&15, K halves [(lane>>4)*16 .. +16)
// B-frag per lane: col = lane&15, same K split.  D: lane&15 = col, VGPR v -> M = v + 8*(lane>>4)
// Keep inner loop simple (round-1 form): compiler hoists the A fragments into
// registers once per tile loop and chains WMMAs; cross-wave parallelism (8 waves)
// hides the B-load latency. A one-tile-ahead prefetch warms L0/L2 for the next
// B stream without adding live VGPRs.
__device__ __forceinline__ void wmma_gemm(const _Float16* sA, int lda,
                                          const _Float16* __restrict__ WT, int K,
                                          float* __restrict__ outBuf, int ldo,
                                          int nTiles, int tid) {
    const int wave  = tid >> 5;
    const int lane  = tid & 31;
    const int row   = lane & 15;
    const int khalf = (lane >> 4) << 4;
    for (int tile = wave; tile < nTiles; tile += 8) {
        const int nb = tile << 4;
        v8f acc = {};
        const _Float16* arow = sA + (size_t)row * lda + khalf;
        const _Float16* wrow = WT + (size_t)(nb + row) * K + khalf;
        if (tile + 8 < nTiles)   // warm next tile's B stream
            __builtin_prefetch(WT + (size_t)(nb + 128 + row) * K + khalf, 0, 1);
        for (int kb = 0; kb < K; kb += 32) {
            v16h a = *(const v16h*)(arow + kb);
            v16h b = *(const v16h*)(wrow + kb);
            acc = __builtin_amdgcn_wmma_f32_16x16x32_f16(
                false, a, false, b, (short)0, acc, false, false);
        }
        const int mbase = (lane >> 4) << 3;
        const int colg  = nb + row;
#pragma unroll
        for (int v = 0; v < 8; ++v)
            outBuf[(size_t)(mbase + v) * ldo + colg] = acc[v];
    }
}

// in-place softmax over 512 LDS floats, 256 threads
__device__ __forceinline__ void softmax512(float* v, float* red, int tid) {
    float mx = -1e30f;
    for (int n = tid; n < N_MEM; n += 256) mx = fmaxf(mx, v[n]);
    red[tid] = mx; __syncthreads();
    for (int s = 128; s > 0; s >>= 1) { if (tid < s) red[tid] = fmaxf(red[tid], red[tid + s]); __syncthreads(); }
    mx = red[0]; __syncthreads();
    float sm = 0.0f;
    for (int n = tid; n < N_MEM; n += 256) { float e = __expf(v[n] - mx); v[n] = e; sm += e; }
    red[tid] = sm; __syncthreads();
    for (int s = 128; s > 0; s >>= 1) { if (tid < s) red[tid] += red[tid + s]; __syncthreads(); }
    float inv = 1.0f / red[0]; __syncthreads();
    for (int n = tid; n < N_MEM; n += 256) v[n] *= inv;
    __syncthreads();
}

// ---------------- main persistent DNC kernel: 2 blocks x 256 threads ----------------
__global__ __launch_bounds__(256) void dnc_main(
    const float* __restrict__ x,        // [T][B][IN]
    const float* __restrict__ b_lstm,   // [1024]
    const float* __restrict__ b_if,     // [471]
    const float* __restrict__ b_out,    // [64]
    const _Float16* __restrict__ WgT,
    const _Float16* __restrict__ WifT,
    const _Float16* __restrict__ WoutT,
    float* __restrict__ mem,            // [B][N][WD]
    float* __restrict__ link,           // [B][N][N]
    float* __restrict__ rwst,           // [B][R][N]
    float* __restrict__ wwst,           // [B][N]
    float* __restrict__ precst,         // [B][N]
    float* __restrict__ usagest,        // [B][N]
    float* __restrict__ hstate,         // [B][H]
    float* __restrict__ cstate,         // [B][H]
    float* __restrict__ readsst,        // [B][R*WD]
    float* __restrict__ gatesbuf,       // [2][16][1024]
    float* __restrict__ itfbuf,         // [2][16][480]
    float* __restrict__ out)            // [T][B][OUT]
{
    const int tid  = threadIdx.x;
    const int blk  = blockIdx.x;
    const int b0   = blk * 16;
    const int wv   = tid >> 5;   // wave id within block (8 waves)
    const int lane = tid & 31;

    __shared__ __align__(32) _Float16 sA[16 * KG];   // 18432 B
    __shared__ float u_s[N_MEM];
    __shared__ float alloc_s[N_MEM];
    __shared__ float norm_s[N_MEM];
    __shared__ float ww_s[N_MEM];
    __shared__ float prec_s[N_MEM];
    __shared__ float rw_s[RH * N_MEM];
    __shared__ float fw_s[RH * N_MEM];
    __shared__ float bw_s[RH * N_MEM];
    __shared__ float rc_s[RH * N_MEM];
    __shared__ float keys_s[RH * WD];
    __shared__ float wkey_s[WD];
    __shared__ float erase_s[WD];
    __shared__ float wv_s[WD];
    __shared__ float red_s[256];
    __shared__ float scal_s[32];
    // scal layout: [0..3]=read_str [4]=wstr [5..8]=free_g [9]=alloc_g [10]=write_g
    //              [11..14]=bwd_mode [15..18]=fwd_mode [19..22]=content_mode
    //              [23..26]=read key norms [27]=write key norm

    // ---- zero-init this block's recurrent state ----
    for (size_t i = tid; i < (size_t)16 * N_MEM * WD; i += 256) mem[(size_t)b0 * N_MEM * WD + i] = 0.0f;
    for (size_t i = tid; i < (size_t)16 * N_MEM * N_MEM; i += 256) link[(size_t)b0 * N_MEM * N_MEM + i] = 0.0f;
    for (size_t i = tid; i < (size_t)16 * RH * N_MEM; i += 256) rwst[(size_t)b0 * RH * N_MEM + i] = 0.0f;
    for (size_t i = tid; i < (size_t)16 * N_MEM; i += 256) {
        wwst[(size_t)b0 * N_MEM + i] = 0.0f;
        precst[(size_t)b0 * N_MEM + i] = 0.0f;
        usagest[(size_t)b0 * N_MEM + i] = 0.0f;
    }
    for (size_t i = tid; i < (size_t)16 * HID; i += 256) {
        hstate[(size_t)b0 * HID + i] = 0.0f;
        cstate[(size_t)b0 * HID + i] = 0.0f;
        readsst[(size_t)b0 * HID + i] = 0.0f;  // R*WD == 256 == HID
    }
    __threadfence_block();
    __syncthreads();

    float* gbuf = gatesbuf + (size_t)blk * 16 * NG;
    float* ibuf = itfbuf   + (size_t)blk * 16 * IF_PAD;

    for (int t = 0; t < T_STEPS; ++t) {
        // ---- phase 1: stage A = [x_t | reads | h] as f16 ----
        for (int idx = tid; idx < 16 * KG; idx += 256) {
            int m = idx / KG, k = idx % KG;
            float v;
            if (k < IN_DIM) {
                v = x[((size_t)t * BATCH + (b0 + m)) * IN_DIM + k];
                if (t + 1 < T_STEPS)   // pull next step's input toward L2/L0
                    __builtin_prefetch(&x[((size_t)(t + 1) * BATCH + (b0 + m)) * IN_DIM + k], 0, 1);
            }
            else if (k < 320)      v = readsst[(size_t)(b0 + m) * 256 + (k - IN_DIM)];
            else                   v = hstate[(size_t)(b0 + m) * HID + (k - 320)];
            sA[idx] = (_Float16)v;
        }
        __syncthreads();
        wmma_gemm(sA, KG, WgT, KG, gbuf, NG, NG / 16, tid);
        __threadfence_block(); __syncthreads();

        // ---- phase 2: LSTM elementwise ----
        for (int idx = tid; idx < 16 * HID; idx += 256) {
            int m = idx >> 8, j = idx & 255;
            const float* g = gbuf + (size_t)m * NG;
            float ig = g[j]        + b_lstm[j];
            float fg = g[256 + j]  + b_lstm[256 + j];
            float gg = g[512 + j]  + b_lstm[512 + j];
            float og = g[768 + j]  + b_lstm[768 + j];
            size_t si = (size_t)(b0 + m) * HID + j;
            float c  = sigf(fg) * cstate[si] + sigf(ig) * tanhf(gg);
            float h  = sigf(og) * tanhf(c);
            hstate[si] = clipf(h);
            cstate[si] = clipf(c);
        }
        __threadfence_block(); __syncthreads();

        // ---- phase 3: interface GEMM (A = hc) ----
        for (int idx = tid; idx < 16 * HID; idx += 256) {
            int m = idx >> 8, k = idx & 255;
            sA[m * KG + k] = (_Float16)hstate[(size_t)(b0 + m) * HID + k];
        }
        __syncthreads();
        wmma_gemm(sA, KG, WifT, HID, ibuf, IF_PAD, IF_PAD / 16, tid);
        __threadfence_block(); __syncthreads();

        // ---- phase 4: per-row memory machinery ----
        for (int m = 0; m < 16; ++m) {
            const int b = b0 + m;
            const float* itf = ibuf + (size_t)m * IF_PAD;
            float* memb  = mem  + (size_t)b * N_MEM * WD;
            float* linkb = link + (size_t)b * N_MEM * N_MEM;

            // P1: parse interface vector (with bias), load rw/prec
            for (int i = tid; i < RH * WD; i += 256) keys_s[i] = itf[i] + b_if[i];
            if (tid < WD) {
                wkey_s[tid]  = itf[260 + tid] + b_if[260 + tid];
                erase_s[tid] = sigf(itf[325 + tid] + b_if[325 + tid]);
                wv_s[tid]    = itf[389 + tid] + b_if[389 + tid];
            }
            if (tid < 4)  scal_s[tid] = oneplusf(itf[256 + tid] + b_if[256 + tid]);       // read_str
            if (tid == 4) scal_s[4]   = oneplusf(itf[324] + b_if[324]);                   // write_str
            if (tid >= 5 && tid < 9) scal_s[tid] = sigf(itf[453 + tid - 5] + b_if[453 + tid - 5]); // free_g
            if (tid == 9)  scal_s[9]  = sigf(itf[457] + b_if[457]);                       // alloc_g
            if (tid == 10) scal_s[10] = sigf(itf[458] + b_if[458]);                       // write_g
            if (tid < 4) { // modes softmax over 3, r = tid
                float a0 = itf[459 + tid * 3 + 0] + b_if[459 + tid * 3 + 0];
                float a1 = itf[459 + tid * 3 + 1] + b_if[459 + tid * 3 + 1];
                float a2 = itf[459 + tid * 3 + 2] + b_if[459 + tid * 3 + 2];
                float mx = fmaxf(a0, fmaxf(a1, a2));
                float e0 = __expf(a0 - mx), e1 = __expf(a1 - mx), e2 = __expf(a2 - mx);
                float s = e0 + e1 + e2;
                scal_s[11 + tid] = e0 / s;   // bwd
                scal_s[15 + tid] = e1 / s;   // fwd
                scal_s[19 + tid] = e2 / s;   // content
            }
            for (int i = tid; i < RH * N_MEM; i += 256) rw_s[i]   = rwst[(size_t)b * RH * N_MEM + i];
            for (int i = tid; i < N_MEM; i += 256)      prec_s[i] = precst[(size_t)b * N_MEM + i];
            __syncthreads();

            // P2: usage update, memory norms (pre-write), key norms
            for (int n = tid; n < N_MEM; n += 256) {
                float us = usagest[(size_t)b * N_MEM + n];
                float wwp = wwst[(size_t)b * N_MEM + n];
                float u = us + (1.0f - us) * wwp;
                float psi = 1.0f;
#pragma unroll
                for (int r = 0; r < RH; ++r) psi *= (1.0f - scal_s[5 + r] * rw_s[r * N_MEM + n]);
                float un = u * psi;
                usagest[(size_t)b * N_MEM + n] = un;
                u_s[n] = EPSV + (1.0f - EPSV) * un;
            }
            for (int n = tid; n < N_MEM; n += 256) {
                const float* mr = memb + (size_t)n * WD;
                float s = 0.0f;
                for (int w = 0; w < WD; ++w) s += mr[w] * mr[w];
                norm_s[n] = sqrtf(s);
            }
            if (tid < 4) {
                float s = 0.0f;
                for (int w = 0; w < WD; ++w) { float k = keys_s[tid * WD + w]; s += k * k; }
                scal_s[23 + tid] = sqrtf(s);
            }
            if (tid == 4) {
                float s = 0.0f;
                for (int w = 0; w < WD; ++w) s += wkey_s[w] * wkey_s[w];
                scal_s[27] = sqrtf(s);
            }
            __syncthreads();

            // P3: allocation (sort-free, stable-argsort semantics) + write-content sim (into fw_s[0..511])
            for (int i = tid; i < N_MEM; i += 256) {
                float ui = u_s[i];
                float prod = 1.0f;
                for (int j = 0; j < N_MEM; ++j) {
                    float uj = u_s[j];
                    bool less = (uj < ui) || (uj == ui && j < i);
                    prod *= less ? uj : 1.0f;
                }
                alloc_s[i] = (1.0f - ui) * prod;
            }
            for (int n = tid; n < N_MEM; n += 256) {
                const float* mr = memb + (size_t)n * WD;
                float d = 0.0f;
                for (int w = 0; w < WD; ++w) d += wkey_s[w] * mr[w];
                fw_s[n] = scal_s[4] * d / (scal_s[27] * norm_s[n] + EPSV);
            }
            __syncthreads();
            softmax512(fw_s, red_s, tid);   // fw_s[0..511] = write content weights

            // P4: write weights
            {
                float ag = scal_s[9], wg = scal_s[10];
                for (int n = tid; n < N_MEM; n += 256) {
                    float w = wg * (ag * alloc_s[n] + (1.0f - ag) * fw_s[n]);
                    ww_s[n] = w;
                    wwst[(size_t)b * N_MEM + n] = w;
                }
            }
            __syncthreads();

            // P5: memory write
            for (int idx = tid; idx < N_MEM * WD; idx += 256) {
                int n = idx >> 6, w = idx & 63;
                float wwn = ww_s[n];
                memb[idx] = memb[idx] * (1.0f - wwn * erase_s[w]) + wwn * wv_s[w];
            }
            __syncthreads();

            // P6: link update (one wave per row, lanes coalesced over j) fused with fw
            for (int i = wv; i < N_MEM; i += 8) {
                float wwi = ww_s[i];
                float f0 = 0, f1 = 0, f2 = 0, f3 = 0;
                float* lrow = linkb + (size_t)i * N_MEM;
                for (int j = lane; j < N_MEM; j += 32) {
                    float nl = (1.0f - wwi - ww_s[j]) * lrow[j] + wwi * prec_s[j];
                    nl = (i == j) ? 0.0f : nl;
                    lrow[j] = nl;
                    f0 += nl * rw_s[j];
                    f1 += nl * rw_s[N_MEM + j];
                    f2 += nl * rw_s[2 * N_MEM + j];
                    f3 += nl * rw_s[3 * N_MEM + j];
                }
#pragma unroll
                for (int off = 16; off > 0; off >>= 1) {
                    f0 += __shfl_xor(f0, off, 32);
                    f1 += __shfl_xor(f1, off, 32);
                    f2 += __shfl_xor(f2, off, 32);
                    f3 += __shfl_xor(f3, off, 32);
                }
                if (lane == 0) {
                    fw_s[i] = f0; fw_s[N_MEM + i] = f1;
                    fw_s[2 * N_MEM + i] = f2; fw_s[3 * N_MEM + i] = f3;
                }
            }
            __threadfence_block();

            // P7: precedence update (needs sum of ww)
            {
                float s = 0.0f;
                for (int n = tid; n < N_MEM; n += 256) s += ww_s[n];
                red_s[tid] = s; __syncthreads();
                for (int st = 128; st > 0; st >>= 1) { if (tid < st) red_s[tid] += red_s[tid + st]; __syncthreads(); }
                float wwsum = red_s[0]; __syncthreads();
                for (int n = tid; n < N_MEM; n += 256)
                    precst[(size_t)b * N_MEM + n] = (1.0f - wwsum) * prec_s[n] + ww_s[n];
            }
            __syncthreads();

            // P8: read content weights on updated memory
            for (int n = tid; n < N_MEM; n += 256) {
                const float* mr = memb + (size_t)n * WD;
                float s = 0.0f;
                for (int w = 0; w < WD; ++w) s += mr[w] * mr[w];
                norm_s[n] = sqrtf(s);
            }
            __syncthreads();
            for (int n = tid; n < N_MEM; n += 256) {
                const float* mr = memb + (size_t)n * WD;
                float d0 = 0, d1 = 0, d2 = 0, d3 = 0;
                for (int w = 0; w < WD; ++w) {
                    float mv = mr[w];
                    d0 += keys_s[w] * mv;
                    d1 += keys_s[WD + w] * mv;
                    d2 += keys_s[2 * WD + w] * mv;
                    d3 += keys_s[3 * WD + w] * mv;
                }
                float inv;
                inv = 1.0f / (scal_s[23] * norm_s[n] + EPSV); rc_s[n]             = scal_s[0] * d0 * inv;
                inv = 1.0f / (scal_s[24] * norm_s[n] + EPSV); rc_s[N_MEM + n]     = scal_s[1] * d1 * inv;
                inv = 1.0f / (scal_s[25] * norm_s[n] + EPSV); rc_s[2 * N_MEM + n] = scal_s[2] * d2 * inv;
                inv = 1.0f / (scal_s[26] * norm_s[n] + EPSV); rc_s[3 * N_MEM + n] = scal_s[3] * d3 * inv;
            }
            __syncthreads();
            for (int r = 0; r < RH; ++r) softmax512(rc_s + r * N_MEM, red_s, tid);

            // P9: backward weights (column pass over updated link; coalesced across threads)
            for (int i = tid; i < N_MEM; i += 256) {
                float a0 = 0, a1 = 0, a2 = 0, a3 = 0;
                for (int j = 0; j < N_MEM; ++j) {
                    float lv = linkb[(size_t)j * N_MEM + i];
                    a0 += lv * rw_s[j];
                    a1 += lv * rw_s[N_MEM + j];
                    a2 += lv * rw_s[2 * N_MEM + j];
                    a3 += lv * rw_s[3 * N_MEM + j];
                }
                bw_s[i] = a0; bw_s[N_MEM + i] = a1; bw_s[2 * N_MEM + i] = a2; bw_s[3 * N_MEM + i] = a3;
            }
            __syncthreads();

            // P10: read weight update
            for (int n = tid; n < N_MEM; n += 256) {
#pragma unroll
                for (int r = 0; r < RH; ++r) {
                    float rn = scal_s[19 + r] * rc_s[r * N_MEM + n]
                             + scal_s[15 + r] * fw_s[r * N_MEM + n]
                             + scal_s[11 + r] * bw_s[r * N_MEM + n];
                    rw_s[r * N_MEM + n] = rn;
                    rwst[(size_t)b * RH * N_MEM + r * N_MEM + n] = rn;
                }
            }
            __syncthreads();

            // P11: reads = rw @ mem  (tid -> (r,w), exactly 256 threads)
            {
                int r = tid >> 6, w = tid & 63;
                float acc = 0.0f;
                for (int n = 0; n < N_MEM; ++n) acc += rw_s[r * N_MEM + n] * memb[(size_t)n * WD + w];
                readsst[(size_t)b * 256 + r * WD + w] = acc;
            }
            __threadfence_block(); __syncthreads();
        } // row loop

        // ---- phase 5: output GEMM  A = [hc | reads] ----
        for (int idx = tid; idx < 16 * 512; idx += 256) {
            int m = idx >> 9, k = idx & 511;
            float v = (k < HID) ? hstate[(size_t)(b0 + m) * HID + k]
                                : readsst[(size_t)(b0 + m) * 256 + (k - HID)];
            sA[m * KG + k] = (_Float16)v;
        }
        __syncthreads();
        wmma_gemm(sA, KG, WoutT, 512, gbuf, OUT_DIM, OUT_DIM / 16, tid);
        __threadfence_block(); __syncthreads();
        for (int idx = tid; idx < 16 * OUT_DIM; idx += 256) {
            int m = idx >> 6, o = idx & 63;
            float v = gbuf[(size_t)m * OUT_DIM + o] + b_out[o];
            out[((size_t)t * BATCH + (b0 + m)) * OUT_DIM + o] = clipf(v);
        }
        __threadfence_block(); __syncthreads();
    } // time loop
}

// ---------------- host launch ----------------
extern "C" void kernel_launch(void* const* d_in, const int* in_sizes, int n_in,
                              void* d_out, int out_size, void* d_ws, size_t ws_size,
                              hipStream_t stream) {
    const float* x      = (const float*)d_in[0];
    const float* Wx     = (const float*)d_in[1];
    const float* Wh     = (const float*)d_in[2];
    const float* b_lstm = (const float*)d_in[3];
    const float* W_if   = (const float*)d_in[4];
    const float* b_if   = (const float*)d_in[5];
    const float* W_out  = (const float*)d_in[6];
    const float* b_out  = (const float*)d_in[7];

    char* ws = (char*)d_ws;
    size_t off = 0;
    auto carve = [&](size_t bytes) -> void* {
        void* p = ws + off;
        off = (off + bytes + 255) & ~(size_t)255;
        return p;
    };
    _Float16* WgT    = (_Float16*)carve((size_t)NG * KG * 2);
    _Float16* WifT   = (_Float16*)carve((size_t)IF_PAD * HID * 2);
    _Float16* WoutT  = (_Float16*)carve((size_t)OUT_DIM * 512 * 2);
    float* mem       = (float*)carve((size_t)BATCH * N_MEM * WD * 4);
    float* link      = (float*)carve((size_t)BATCH * N_MEM * N_MEM * 4);
    float* rwst      = (float*)carve((size_t)BATCH * RH * N_MEM * 4);
    float* wwst      = (float*)carve((size_t)BATCH * N_MEM * 4);
    float* precst    = (float*)carve((size_t)BATCH * N_MEM * 4);
    float* usagest   = (float*)carve((size_t)BATCH * N_MEM * 4);
    float* hstate    = (float*)carve((size_t)BATCH * HID * 4);
    float* cstate    = (float*)carve((size_t)BATCH * HID * 4);
    float* readsst   = (float*)carve((size_t)BATCH * 256 * 4);
    float* gatesbuf  = (float*)carve((size_t)2 * 16 * NG * 4);
    float* itfbuf    = (float*)carve((size_t)2 * 16 * IF_PAD * 4);
    (void)ws_size; (void)n_in; (void)in_sizes; (void)out_size;

    const int totalConv = NG * KG + IF_PAD * HID + OUT_DIM * 512;
    convert_weights<<<(totalConv + 255) / 256, 256, 0, stream>>>(
        Wx, Wh, W_if, W_out, WgT, WifT, WoutT);

    dnc_main<<<2, 256, 0, stream>>>(
        x, b_lstm, b_if, b_out, WgT, WifT, WoutT,
        mem, link, rwst, wwst, precst, usagest,
        hstate, cstate, readsst, gatesbuf, itfbuf, (float*)d_out);
}
